// NaiveLSTM_16269336118036
// MI455X (gfx1250) — matrix operand order repaired and checked by reading, not compiled, not run on hardware
//
#include <hip/hip_runtime.h>
#include <hip/hip_bf16.h>
#include <math.h>

// ---------------------------------------------------------------------------
// LSTM on MI455X (gfx1250, wave32, WMMA), fragment-precomputed + pipelined.
//
//   x:   [B=64, T=512, IN=512]  fp32
//   W_*: [1536, 1024] fp32, b_*: [1024] fp32  (gates i, f, o, c)
//   out: hs [T*B, 1024] ++ h_T [64,1024] ++ c_T [64,1024]   (fp32)
//
// Pre-passes (once per call): weights -> f16 B-fragment layout (12.6 MB,
// L2-resident), x -> f16 A-fragment layout (33.5 MB).
// Step kernel (512 serial launches): one wave32 owns one 16x16 tile and all
// four gates; inner loop is software-pipelined (preload kt+1 fragments while
// kt's 4 WMMAs execute). Cell update is register-local; c-state is kept in
// accumulator-tile layout so lane accesses are contiguous 32B.
// ---------------------------------------------------------------------------

#define BATCH 64
#define TLEN  512
#define INSZ  512
#define HID   1024
#define KTOT  1536
#define NKT   48          // KTOT / 32
#define NKT_X 16          // INSZ / 32
#define NKT_H 32          // HID / 32

typedef __attribute__((ext_vector_type(16))) _Float16 v16h;
typedef __attribute__((ext_vector_type(8)))  float    v8f;

__device__ __forceinline__ float fast_sigmoid(float x) {
    return 1.0f / (1.0f + __expf(-x));
}
__device__ __forceinline__ float fast_tanh(float x) {
    return 1.0f - 2.0f / (__expf(2.0f * x) + 1.0f);
}

__global__ void lstm_init_state(float* __restrict__ p, int n) {
    int i = blockIdx.x * blockDim.x + threadIdx.x;
    if (i < n) p[i] = 0.0f;
}

// ---- weights -> f16 B-fragment layout -------------------------------------
// wfrag half-index: (((g*64 + cb)*48 + kt)*32 + lane)*16 + slot
//   B 32x16 layout: lane holds N = lane&15, K = (slot&7) + 8*(lane>>4) + 16*(slot>>3)
__global__ void lstm_conv_w(const float* __restrict__ Wi, const float* __restrict__ Wf,
                            const float* __restrict__ Wo, const float* __restrict__ Wc,
                            _Float16* __restrict__ wfrag, int n) {
    int idx = blockIdx.x * blockDim.x + threadIdx.x;
    if (idx >= n) return;
    const int slot = idx & 15;
    const int lane = (idx >> 4) & 31;
    const int t2   = idx >> 9;
    const int kt   = t2 % NKT;
    const int t3   = t2 / NKT;
    const int cb   = t3 & 63;
    const int g    = t3 >> 6;
    const float* Wg = (g == 0) ? Wi : (g == 1) ? Wf : (g == 2) ? Wo : Wc;
    const int nn  = lane & 15;
    const int kk  = (slot & 7) + 8 * (lane >> 4) + 16 * (slot >> 3);
    const int k   = kt * 32 + kk;
    const int col = cb * 16 + nn;
    wfrag[idx] = (_Float16)Wg[(size_t)k * HID + col];
}

// ---- x -> f16 A-fragment layout -------------------------------------------
// xfrag half-index: (((t*4 + rb)*16 + kt)*32 + lane)*16 + slot
//   A 16x32 layout: lane holds M = lane&15; lane half selects K 0..7 vs 8..15
//   (slots 0..7) and K 16..23 vs 24..31 (slots 8..15).
__global__ void lstm_conv_x(const float* __restrict__ x,
                            _Float16* __restrict__ xfrag, int n) {
    int idx = blockIdx.x * blockDim.x + threadIdx.x;
    if (idx >= n) return;
    const int slot = idx & 15;
    const int lane = (idx >> 4) & 31;
    const int kt   = (idx >> 9) & 15;
    const int rb   = (idx >> 13) & 3;
    const int t    = idx >> 15;
    const int m  = lane & 15;
    const int hb = lane >> 4;
    const int sb = slot >> 3;
    const int kk = (slot & 7) + 8 * hb + 16 * sb;
    const int b  = rb * 16 + m;
    const int k  = kt * 32 + kk;
    xfrag[idx] = (_Float16)x[(size_t)b * (TLEN * INSZ) + (size_t)t * INSZ + k];
}

#define WMMA4(AF, I)                                                                        \
    a0 = __builtin_amdgcn_wmma_f32_16x16x32_f16(false, AF, false, b0, (short)0, a0, false, false); \
    a1 = __builtin_amdgcn_wmma_f32_16x16x32_f16(false, AF, false, b1, (short)0, a1, false, false); \
    a2 = __builtin_amdgcn_wmma_f32_16x16x32_f16(false, AF, false, b2, (short)0, a2, false, false); \
    a3 = __builtin_amdgcn_wmma_f32_16x16x32_f16(false, AF, false, b3, (short)0, a3, false, false);

// ---- one LSTM timestep: 256 waves, one 16x16 tile x 4 gates per wave ------
__global__ __launch_bounds__(128)
void lstm_step(const _Float16* __restrict__ xfrag,
               const _Float16* __restrict__ wfrag,
               const _Float16* __restrict__ hp,    // h_{t-1}, A-frag layout
               _Float16*       __restrict__ hn,    // h_t, A-frag layout
               const float* __restrict__ bi, const float* __restrict__ bf,
               const float* __restrict__ bo, const float* __restrict__ bc,
               float* __restrict__ cbuf,           // tile-layout c state
               float* __restrict__ out,
               int t)
{
    const int tid  = threadIdx.x;
    const int lane = tid & 31;
    const int gw   = blockIdx.x * 4 + (tid >> 5);   // 0..255 global wave id
    const int rb   = gw & 3;                        // batch row block
    const int cb   = gw >> 2;                       // hidden col block
    const int hh   = lane >> 4;
    const int l16  = lane & 15;

    // fragment stream bases (v16h = 32B per lane per K-tile)
    const v16h* aX = (const v16h*)xfrag + ((size_t)(t * 4 + rb) * NKT_X) * 32 + lane;
    const v16h* aH = (const v16h*)hp    + ((size_t)rb * NKT_H) * 32 + lane;
    const v16h* w0 = (const v16h*)wfrag + ((size_t)(0 * 64 + cb) * NKT) * 32 + lane;
    const v16h* w1 = (const v16h*)wfrag + ((size_t)(1 * 64 + cb) * NKT) * 32 + lane;
    const v16h* w2 = (const v16h*)wfrag + ((size_t)(2 * 64 + cb) * NKT) * 32 + lane;
    const v16h* w3 = (const v16h*)wfrag + ((size_t)(3 * 64 + cb) * NKT) * 32 + lane;

    __builtin_prefetch(aH, 0, 1);   // warm recurrent-h stream (global_prefetch_b8)

    v8f a0 = {}, a1 = {}, a2 = {}, a3 = {};

    // ---- software-pipelined K loop: preload kt+1 while kt computes --------
    v16h af = aX[0];
    v16h b0 = w0[0], b1 = w1[0], b2 = w2[0], b3 = w3[0];

    // K 0..511: x contribution
    #pragma unroll
    for (int kt = 0; kt < NKT_X; ++kt) {
        const size_t nw = (size_t)(kt + 1) * 32;            // next weight tile
        const v16h afn = (kt + 1 < NKT_X) ? aX[nw] : aH[0]; // next A fragment
        const v16h b0n = w0[nw];
        const v16h b1n = w1[nw];
        const v16h b2n = w2[nw];
        const v16h b3n = w3[nw];
        WMMA4(af, kt)
        af = afn; b0 = b0n; b1 = b1n; b2 = b2n; b3 = b3n;
    }
    // K 512..1535: recurrent h contribution
    #pragma unroll
    for (int kc = 0; kc < NKT_H; ++kc) {
        const int ktn = NKT_X + ((kc + 1 < NKT_H) ? kc + 1 : kc);  // clamped
        const size_t nw = (size_t)ktn * 32;
        const v16h afn = aH[(size_t)(ktn - NKT_X) * 32];
        const v16h b0n = w0[nw];
        const v16h b1n = w1[nw];
        const v16h b2n = w2[nw];
        const v16h b3n = w3[nw];
        WMMA4(af, kc)
        af = afn; b0 = b0n; b1 = b1n; b2 = b2n; b3 = b3n;
    }

    // ---- register-local LSTM cell update ----------------------------------
    // C/D layout: VGPR j of each accumulator is element (M = hh*8 + j, N = l16)
    // for ALL FOUR gates simultaneously.
    const int col = cb * 16 + l16;
    const float Bi = bi[col], Bf = bf[col], Bo = bo[col], Bc = bc[col];

    // c-state in accumulator-tile layout: 32B contiguous per lane
    float* cw = cbuf + (size_t)gw * 256 + lane * 8;

    const int kc2   = cb >> 1;                     // h-fragment K-tile index
    const int lane2 = 16 * (l16 >> 3);             // + m below
    const int slot2 = (l16 & 7) + ((cb & 1) << 3);

    #pragma unroll
    for (int j = 0; j < 8; ++j) {
        const float iv = fast_sigmoid(a0[j] + Bi);
        const float fv = fast_sigmoid(a1[j] + Bf);
        const float ov = fast_sigmoid(a2[j] + Bo);
        const float cv = fast_tanh   (a3[j] + Bc);

        const float c_new = fv * cw[j] + iv * cv;
        cw[j] = c_new;
        const float h_new = ov * fast_tanh(c_new);

        const int m = hh * 8 + j;
        const int b = rb * 16 + m;
        out[((size_t)t * BATCH + b) * HID + col] = h_new;

        // h_t in A-fragment layout for step t+1
        hn[((size_t)(rb * NKT_H + kc2) * 32 + (m + lane2)) * 16 + slot2] = (_Float16)h_new;

        if (t == TLEN - 1) {
            const size_t hs_sz = (size_t)TLEN * BATCH * HID;
            const size_t cidx  = (size_t)b * HID + col;
            out[hs_sz + cidx]               = h_new;   // h_T
            out[hs_sz + BATCH * HID + cidx] = c_new;   // c_T
        }
    }
}

extern "C" void kernel_launch(void* const* d_in, const int* in_sizes, int n_in,
                              void* d_out, int out_size, void* d_ws, size_t ws_size,
                              hipStream_t stream) {
    (void)in_sizes; (void)n_in; (void)out_size; (void)ws_size;

    const float* x  = (const float*)d_in[0];
    const float* Wi = (const float*)d_in[1];
    const float* bi = (const float*)d_in[2];
    const float* Wf = (const float*)d_in[3];
    const float* bf = (const float*)d_in[4];
    const float* Wo = (const float*)d_in[5];
    const float* bo = (const float*)d_in[6];
    const float* Wc = (const float*)d_in[7];
    const float* bc = (const float*)d_in[8];
    float* out = (float*)d_out;

    // ---- workspace layout ----
    // [cbuf: 64K f32][hfrag0: 64K f16][hfrag1: 64K f16][wfrag: 6.29M f16][xfrag: 16.7M f16]
    float*    cbuf   = (float*)d_ws;
    _Float16* hfrag0 = (_Float16*)(cbuf + BATCH * HID);
    _Float16* hfrag1 = hfrag0 + BATCH * HID;
    _Float16* wfrag  = hfrag1 + BATCH * HID;
    _Float16* xfrag  = wfrag + (size_t)4 * 64 * NKT * 32 * 16;

    // zero c-state + hfrag0 (contiguous: 64K f32 + 64K f16 = 96K f32 worth)
    const int zn = BATCH * HID + (BATCH * HID) / 2;
    lstm_init_state<<<(zn + 255) / 256, 256, 0, stream>>>(cbuf, zn);

    const int wn = 4 * 64 * NKT * 32 * 16;          // 6,291,456
    lstm_conv_w<<<(wn + 255) / 256, 256, 0, stream>>>(Wi, Wf, Wo, Wc, wfrag, wn);

    const int xn = TLEN * 4 * NKT_X * 32 * 16;      // 16,777,216
    lstm_conv_x<<<(xn + 255) / 256, 256, 0, stream>>>(x, xfrag, xn);

    for (int t = 0; t < TLEN; ++t) {
        const _Float16* hp = (t & 1) ? hfrag1 : hfrag0;
        _Float16*       hn = (t & 1) ? hfrag0 : hfrag1;
        lstm_step<<<64, 128, 0, stream>>>(xfrag, wfrag, hp, hn,
                                          bi, bf, bo, bc, cbuf, out, t);
    }
}
